// SelfAttention_v1_v2_90847148245684
// MI455X (gfx1250) — matrix operand order repaired and checked by reading, not compiled, not run on hardware
//
#include <hip/hip_runtime.h>

typedef __bf16 bf16;
typedef __attribute__((ext_vector_type(8)))  bf16  v8bf;
typedef __attribute__((ext_vector_type(16))) bf16  v16bf;
typedef __attribute__((ext_vector_type(8)))  float v8f;

#define SEQ 4096
#define DIM 1024
// (1/sqrt(1024)) * log2(e) : fold softmax scale into the exp2 exponent
#define SCALE_LOG2E 0.045084220027780106f

static __device__ __forceinline__ v8f splat8(float x) {
  v8f r;
#pragma unroll
  for (int i = 0; i < 8; ++i) r[i] = x;
  return r;
}

// ---- WMMA fragment loaders (bf16, 16x16x32) -------------------------------
// A (16x32, row-major source, lane holds row m = lane&15):
//   VGPR0-3 : K = 8*half + 0..7        (16B run at k0 + 8*half)
//   VGPR4-7 : K = 16 + 8*half + 0..7   (16B run at k0 + 16 + 8*half)
static __device__ __forceinline__ v16bf ld_afrag(const bf16* base, int ld, int k0) {
  const int lane = threadIdx.x & 31;
  const bf16* p = base + (lane & 15) * ld + k0 + 8 * (lane >> 4);
  v8bf lo = *(const v8bf*)p;
  v8bf hi = *(const v8bf*)(p + 16);
  return __builtin_shufflevector(lo, hi, 0,1,2,3,4,5,6,7,8,9,10,11,12,13,14,15);
}

// B (32x16) from N-major storage (row n holds its K-run; lane holds n = lane&15):
//   VGPR0-3 : K = 8*half + 0..7        (16B run at k0 + 8*half)
//   VGPR4-7 : K = 16 + 8*half + 0..7   (16B run at k0 + 16 + 8*half)
static __device__ __forceinline__ v16bf ld_bfrag(const bf16* base, int ld, int k0) {
  const int lane = threadIdx.x & 31;
  const bf16* p = base + (lane & 15) * ld + k0 + 8 * (lane >> 4);
  v8bf lo = *(const v8bf*)p;
  v8bf hi = *(const v8bf*)(p + 16);
  return __builtin_shufflevector(lo, hi, 0,1,2,3,4,5,6,7,8,9,10,11,12,13,14,15);
}

// ---- async HBM -> LDS (GLOBAL_LOAD_ASYNC_TO_LDS_B128, ASYNCcnt) -----------
// Generic pointer to __shared__ carries the LDS offset in its low 32 bits.
static __device__ __forceinline__ void async_ld_b128(bf16* dst_lds, const bf16* src) {
  const uint32_t la = (uint32_t)(uintptr_t)dst_lds;
  const uint64_t ga = (uint64_t)(uintptr_t)src;
  asm volatile("global_load_async_to_lds_b128 %0, %1, off"
               :: "v"(la), "v"(ga) : "memory");
}

// contiguous block copy, split over `nthreads` threads (16B packets)
static __device__ __forceinline__ void async_copy_to_lds(bf16* dst_lds, const bf16* src,
                                                         int n_elems, int tid, int nthreads) {
  const int nv = n_elems / 8;
  for (int i = tid; i < nv; i += nthreads)
    async_ld_b128(dst_lds + i * 8, src + i * 8);
}

// strided 2D copy: `rows` rows of `cols` bf16 from row-major src (ld_src) into
// a packed LDS slab (row stride = cols), split over `nthreads` threads
static __device__ __forceinline__ void async_copy_2d_to_lds(bf16* dst_lds, const bf16* src,
                                                            int rows, int cols, int ld_src,
                                                            int tid, int nthreads) {
  const int pk_per_row = cols / 8;
  const int nv = rows * pk_per_row;
  for (int i = tid; i < nv; i += nthreads) {
    const int r = i / pk_per_row, c = i % pk_per_row;
    async_ld_b128(dst_lds + r * cols + c * 8, src + r * ld_src + c * 8);
  }
}

// ---- stage 1: f32 -> bf16 --------------------------------------------------
__global__ void cvt_bf16_kernel(const float* __restrict__ src, bf16* __restrict__ dst, int n) {
  int i = blockIdx.x * blockDim.x + threadIdx.x;
  if (i < n) dst[i] = (bf16)src[i];
}

// ---- stage 2: QKV projection out = x @ W^T --------------------------------
// grid: (256 m-tiles, 4 n-chunks of 256, 3 matrices), block: 128 (4 waves)
// All WMMA operands come from LDS; W is consumed in 256x128 chunks that are
// async-prefetched one chunk ahead (double buffered).
__global__ __launch_bounds__(128) void qkv_kernel(
    const bf16* __restrict__ xb,
    const bf16* __restrict__ wq, const bf16* __restrict__ wk, const bf16* __restrict__ wv,
    bf16* __restrict__ qb, bf16* __restrict__ kb, bf16* __restrict__ vtb) {
  __shared__ __align__(16) bf16 xs[16 * DIM];        //  32 KB
  __shared__ __align__(16) bf16 wsl[2][256 * 128];   // 128 KB (double buffer)

  const int z = blockIdx.z;
  const bf16* W = (z == 0) ? wq : (z == 1) ? wk : wv;
  const int tid = threadIdx.x;
  const int wave = tid >> 5;
  const int lane = tid & 31;
  const int n = lane & 15, h = lane >> 4;
  const int i0 = blockIdx.x * 16;
  const int nbb = blockIdx.y * 256;           // block's W-row base
  const int nbase = nbb + wave * 64;          // this wave's output-column base

  async_copy_to_lds(xs, xb + i0 * DIM, 16 * DIM, tid, 128);
  async_copy_2d_to_lds(wsl[0], W + nbb * DIM, 256, 128, DIM, tid, 128);
  asm volatile("s_wait_asynccnt 0x0" ::: "memory");
  __syncthreads();

  v8f acc[4];
#pragma unroll
  for (int t = 0; t < 4; ++t) acc[t] = splat8(0.0f);

  int buf = 0;
  for (int kc = 0; kc < DIM; kc += 128) {
    if (kc + 128 < DIM)   // prefetch next W chunk; overlaps the 16 WMMAs below
      async_copy_2d_to_lds(wsl[buf ^ 1], W + nbb * DIM + kc + 128, 256, 128, DIM, tid, 128);

    const bf16* wb = wsl[buf] + wave * 64 * 128;
#pragma unroll
    for (int kk = 0; kk < 128; kk += 32) {
      v16bf a = ld_afrag(xs, DIM, kc + kk);
#pragma unroll
      for (int t = 0; t < 4; ++t) {
        v16bf b = ld_bfrag(wb + t * 16 * 128, 128, kk);
        acc[t] = __builtin_amdgcn_wmma_f32_16x16x32_bf16(false, a, false, b,
                                                         (short)0, acc[t], false, false);
      }
    }
    asm volatile("s_wait_asynccnt 0x0" ::: "memory");
    __syncthreads();
    buf ^= 1;
  }

#pragma unroll
  for (int t = 0; t < 4; ++t) {
    const int col = nbase + t * 16 + n;
#pragma unroll
    for (int v = 0; v < 8; ++v) {
      const int row = i0 + v + 8 * h;
      const bf16 val = (bf16)acc[t][v];
      if (z == 2)      vtb[col * SEQ + row] = val;   // V transposed
      else if (z == 0) qb[row * DIM + col] = val;
      else             kb[row * DIM + col] = val;
    }
  }
}

// ---- stage 3: flash attention ---------------------------------------------
// grid: 256 query blocks of 16 rows; block: 128 threads (4 waves).
// K chunks (workgroup-shared) and per-wave V slices are double-buffered in
// LDS via async global->LDS copies; prefetch of chunk i+1 overlaps all
// WMMA/softmax work of chunk i. Every WMMA operand is LDS-resident.
__global__ __launch_bounds__(128) void attn_kernel(
    const bf16* __restrict__ Q, const bf16* __restrict__ K,
    const bf16* __restrict__ Vt, float* __restrict__ out) {
  __shared__ __align__(16) bf16 Qs[16 * DIM];          //  32 KB
  __shared__ __align__(16) bf16 Ks[2][32 * DIM];       // 128 KB
  __shared__ __align__(16) bf16 Vs[2][4 * 256 * 32];   // 128 KB (per-wave slices)
  __shared__ __align__(16) bf16 Pl[4][16 * 32];        //   4 KB (per-wave P transpose)

  const int tid = threadIdx.x;
  const int wave = tid >> 5;
  const int lane = tid & 31;
  const int n = lane & 15, h = lane >> 4;
  const int i0 = blockIdx.x * 16;
  const int dbase = wave * 256;               // this wave's output-dim base
  bf16* vslab[2] = { Vs[0] + wave * 256 * 32, Vs[1] + wave * 256 * 32 };

  // async-stage Q block, K chunk 0 (workgroup) and V slice 0 (per wave)
  async_copy_to_lds(Qs, Q + i0 * DIM, 16 * DIM, tid, 128);
  async_copy_to_lds(Ks[0], K, 32 * DIM, tid, 128);
  async_copy_2d_to_lds(vslab[0], Vt + dbase * SEQ, 256, 32, SEQ, lane, 32);
  asm volatile("s_wait_asynccnt 0x0" ::: "memory");
  __syncthreads();

  v8f acc[16];
#pragma unroll
  for (int t = 0; t < 16; ++t) acc[t] = splat8(0.0f);
  v8f mrow = splat8(-1e30f);
  v8f lrow = splat8(0.0f);

  int buf = 0;
  for (int kbp = 0; kbp < SEQ; kbp += 32) {
    if (kbp + 32 < SEQ) {  // prefetch next chunk; overlaps everything below
      async_copy_to_lds(Ks[buf ^ 1], K + (kbp + 32) * DIM, 32 * DIM, tid, 128);
      async_copy_2d_to_lds(vslab[buf ^ 1], Vt + dbase * SEQ + kbp + 32, 256, 32, SEQ, lane, 32);
    }

    const bf16* Kb = Ks[buf];

    // S = Q Kt for two 16x16 key tiles (LDS-fed, pipelined fragments)
    v8f s0 = splat8(0.0f), s1 = splat8(0.0f);
    v16bf a  = ld_afrag(Qs, DIM, 0);
    v16bf b0 = ld_bfrag(Kb, DIM, 0);
    v16bf b1 = ld_bfrag(Kb + 16 * DIM, DIM, 0);
#pragma unroll 4
    for (int k0 = 0; k0 + 32 < DIM; k0 += 32) {
      v16bf an  = ld_afrag(Qs, DIM, k0 + 32);
      v16bf b0n = ld_bfrag(Kb, DIM, k0 + 32);
      v16bf b1n = ld_bfrag(Kb + 16 * DIM, DIM, k0 + 32);
      s0 = __builtin_amdgcn_wmma_f32_16x16x32_bf16(false, a, false, b0, (short)0, s0, false, false);
      s1 = __builtin_amdgcn_wmma_f32_16x16x32_bf16(false, a, false, b1, (short)0, s1, false, false);
      a = an; b0 = b0n; b1 = b1n;
    }
    s0 = __builtin_amdgcn_wmma_f32_16x16x32_bf16(false, a, false, b0, (short)0, s0, false, false);
    s1 = __builtin_amdgcn_wmma_f32_16x16x32_bf16(false, a, false, b1, (short)0, s1, false, false);

    // online softmax (row = v + 8*h lives across the 16 lanes of this half)
    v8f c;
#pragma unroll
    for (int v = 0; v < 8; ++v) {
      float mx = fmaxf(s0[v], s1[v]);
      mx = fmaxf(mx, __shfl_xor(mx, 1, 32));
      mx = fmaxf(mx, __shfl_xor(mx, 2, 32));
      mx = fmaxf(mx, __shfl_xor(mx, 4, 32));
      mx = fmaxf(mx, __shfl_xor(mx, 8, 32));
      const float mn = fmaxf(mrow[v], mx);
      const float cc = __builtin_amdgcn_exp2f((mrow[v] - mn) * SCALE_LOG2E);
      const float p0 = __builtin_amdgcn_exp2f((s0[v] - mn) * SCALE_LOG2E);
      const float p1 = __builtin_amdgcn_exp2f((s1[v] - mn) * SCALE_LOG2E);
      float rs = p0 + p1;
      rs += __shfl_xor(rs, 1, 32);
      rs += __shfl_xor(rs, 2, 32);
      rs += __shfl_xor(rs, 4, 32);
      rs += __shfl_xor(rs, 8, 32);
      lrow[v] = cc * lrow[v] + rs;
      mrow[v] = mn;
      c[v] = cc;
      // spill P (C-layout) to LDS so we can re-read it in A-layout
      Pl[wave][(v + 8 * h) * 32 + n]      = (bf16)p0;
      Pl[wave][(v + 8 * h) * 32 + 16 + n] = (bf16)p1;
    }
    asm volatile("s_wait_dscnt 0x0" ::: "memory");
    v16bf pa = ld_afrag(&Pl[wave][0], 32, 0);   // P as 16x32 A fragment

    // O = c*O + P @ V   (B fragments from this wave's LDS V slice)
    const bf16* vb = vslab[buf];
    v16bf bv = ld_bfrag(vb, 32, 0);
#pragma unroll
    for (int t = 0; t < 15; ++t) {
      v16bf bvn = ld_bfrag(vb + (t + 1) * 16 * 32, 32, 0);
#pragma unroll
      for (int v = 0; v < 8; ++v) acc[t][v] *= c[v];
      acc[t] = __builtin_amdgcn_wmma_f32_16x16x32_bf16(false, pa, false, bv,
                                                       (short)0, acc[t], false, false);
      bv = bvn;
    }
#pragma unroll
    for (int v = 0; v < 8; ++v) acc[15][v] *= c[v];
    acc[15] = __builtin_amdgcn_wmma_f32_16x16x32_bf16(false, pa, false, bv,
                                                      (short)0, acc[15], false, false);

    // ensure our prefetch landed, and everyone is done with both buffers
    asm volatile("s_wait_asynccnt 0x0" ::: "memory");
    __syncthreads();
    buf ^= 1;
  }

  // normalize and store f32 output
  v8f invl;
#pragma unroll
  for (int v = 0; v < 8; ++v) invl[v] = 1.0f / lrow[v];
#pragma unroll
  for (int t = 0; t < 16; ++t) {
    const int col = dbase + t * 16 + n;
#pragma unroll
    for (int v = 0; v < 8; ++v)
      out[(i0 + v + 8 * h) * DIM + col] = acc[t][v] * invl[v];
  }
}

extern "C" void kernel_launch(void* const* d_in, const int* in_sizes, int n_in,
                              void* d_out, int out_size, void* d_ws, size_t ws_size,
                              hipStream_t stream) {
  const float* x  = (const float*)d_in[0];
  const float* Wq = (const float*)d_in[1];
  const float* Wk = (const float*)d_in[2];
  const float* Wv = (const float*)d_in[3];

  char* ws = (char*)d_ws;
  bf16* xb  = (bf16*)(ws);                          // 8 MB
  bf16* wqb = (bf16*)(ws + (8u  << 20));            // 2 MB
  bf16* wkb = (bf16*)(ws + (10u << 20));            // 2 MB
  bf16* wvb = (bf16*)(ws + (12u << 20));            // 2 MB
  bf16* qb  = (bf16*)(ws + (14u << 20));            // 8 MB
  bf16* kb  = (bf16*)(ws + (22u << 20));            // 8 MB
  bf16* vtb = (bf16*)(ws + (30u << 20));            // 8 MB (transposed V)

  cvt_bf16_kernel<<<(SEQ * DIM + 255) / 256, 256, 0, stream>>>(x, xb, SEQ * DIM);
  cvt_bf16_kernel<<<(DIM * DIM + 255) / 256, 256, 0, stream>>>(Wq, wqb, DIM * DIM);
  cvt_bf16_kernel<<<(DIM * DIM + 255) / 256, 256, 0, stream>>>(Wk, wkb, DIM * DIM);
  cvt_bf16_kernel<<<(DIM * DIM + 255) / 256, 256, 0, stream>>>(Wv, wvb, DIM * DIM);

  qkv_kernel<<<dim3(SEQ / 16, 4, 3), 128, 0, stream>>>(xb, wqb, wkb, wvb, qb, kb, vtb);
  attn_kernel<<<dim3(SEQ / 16), 128, 0, stream>>>(qb, kb, vtb, (float*)d_out);
}
